// EuclideanCodebook_70909910057264
// MI455X (gfx1250) — matrix operand order repaired and checked by reading, not compiled, not run on hardware
//
#include <hip/hip_runtime.h>
#include <hip/hip_bf16.h>

// ---------------------------------------------------------------------------
// EuclideanCodebook VQ on gfx1250:
//   codes = argmin_k ( e_sq[k] - 2 * x . emb[k] )        (||x||^2 dropped)
//   quantized = emb[codes]
// GEMM via v_wmma_f32_16x16x32_bf16 with 2-term bf16 split of fp32
// (xh*eh + xh*el + xl*eh). B tiles stream through LDS with double-buffered
// global_load_async_to_lds_b128 (ASYNCcnt), no VGPR staging -> no spills.
// ---------------------------------------------------------------------------

#define VQ_EPS 1e-5f

typedef __attribute__((ext_vector_type(16))) __bf16          v16bf;
typedef __attribute__((ext_vector_type(16))) unsigned short  v16u;
typedef __attribute__((ext_vector_type(8)))  unsigned short  v8u;
typedef __attribute__((ext_vector_type(8)))  float           v8f;

union BF16x16 { v16bf bf; v16u u; v8u h[2]; };

static constexpr int D_DIM  = 256;
static constexpr int K_DIM  = 2048;
static constexpr int N_ROWS = 32 * 1024;     // B*T
static constexpr int KT     = K_DIM / 16;    // 128 k-tiles (16 codebook cols each)
static constexpr int DC     = D_DIM / 32;    // 8 inner-dim chunks of 32
// B tile layout: [kt][dc][part(hi,lo)][sub(2)][lane(32)][8 ushorts]
//   -> per-lane ds_load_b128 with 16B lane stride = all 64 LDS banks hit.
//   -> one k-tile = 8*2*2*32*8 ushorts = 16 KB, contiguous per kt.
static constexpr int TILE_U16   = DC * 2 * 2 * 32 * 8;   // 8192
static constexpr int TILE_BYTES = TILE_U16 * 2;          // 16384

static __device__ inline void split_bf16(float x, unsigned short& hi, unsigned short& lo) {
    unsigned int b = __float_as_uint(x);
    hi = (unsigned short)(b >> 16);                       // truncate to bf16
    float fhi = __uint_as_float((unsigned int)hi << 16);
    float r = x - fhi;                                    // residual, ~2^-8 smaller
    lo = (unsigned short)(__float_as_uint(r) >> 16);
}

// ---------------------------------------------------------------------------
// Kernel 1: embedding = esum / clamp(usage, eps); e_sq; bf16 hi/lo scattered
// into the WMMA B-matrix LDS tile layout. One block per code row k.
// ---------------------------------------------------------------------------
__global__ void vq_prep(const float* __restrict__ esum,
                        const float* __restrict__ usage,
                        float* __restrict__ emb,
                        float* __restrict__ esq,
                        unsigned short* __restrict__ btile) {
    int k = blockIdx.x;
    int d = threadIdx.x;
    float u = usage[k];
    u = (u < VQ_EPS) ? VQ_EPS : u;
    float e = esum[(size_t)k * D_DIM + d] / u;
    emb[(size_t)k * D_DIM + d] = e;

    unsigned short hi, lo;
    split_bf16(e, hi, lo);

    // B k order used here (contiguous, same as A): lane = (din>=16)*16 + (k&15),
    // element-in-lane el = din & 15; split into sub = el>>3, pos = el&7.
    int kt   = k >> 4;
    int n    = k & 15;
    int dc   = d >> 5;
    int din  = d & 31;
    int lane = ((din >> 4) << 4) + n;
    int el   = din & 15;
    int sub  = el >> 3;
    int pos  = el & 7;
    size_t tb = (size_t)kt * TILE_U16;
    size_t ih = tb + ((((size_t)(dc * 2 + 0) * 2 + sub) * 32) + lane) * 8 + pos;
    size_t il = tb + ((((size_t)(dc * 2 + 1) * 2 + sub) * 32) + lane) * 8 + pos;
    btile[ih] = hi;
    btile[il] = lo;

    __shared__ float red[256];
    red[d] = e * e;
    __syncthreads();
    for (int s = 128; s > 0; s >>= 1) {
        if (d < s) red[d] += red[d + s];
        __syncthreads();
    }
    if (d == 0) esq[k] = red[0];
}

// ---------------------------------------------------------------------------
// Kernel 2: fused GEMM + argmin. 256 threads = 8 waves, 16 rows per wave,
// 128 rows per WG. B tiles double-buffered in LDS via async DMA.
// ---------------------------------------------------------------------------
__global__ __launch_bounds__(256, 1) void vq_argmin(
        const float* __restrict__ x,
        const unsigned short* __restrict__ btile,
        const float* __restrict__ esq,
        int* __restrict__ codes) {

    __shared__ float s_esq[K_DIM];                                          // 8 KB
    __shared__ __attribute__((aligned(32))) unsigned short s_b[2][TILE_U16]; // 32 KB

    const int tid  = threadIdx.x;
    const int wave = tid >> 5;
    const int lane = tid & 31;
    const int col  = lane & 15;   // C-layout: N = lane&15
    const int half = lane >> 4;   // C-layout: M = j + 8*half

    for (int i = tid; i < K_DIM; i += 256) s_esq[i] = esq[i];

    // ---- async global->LDS tile copy: 16 KB/tile, 64 B per thread (4 x b128)
    const char*    gB      = (const char*)btile + (size_t)tid * 16;
    const unsigned ldsBase = (unsigned)(size_t)(&s_b[0][0]) + (unsigned)tid * 16;
    auto issueTile = [&](int kt, int buf) {
        const char* g = gB + (size_t)kt * TILE_BYTES;
        unsigned    l = ldsBase + (unsigned)buf * TILE_BYTES;
#pragma unroll
        for (int i = 0; i < 4; ++i) {
            asm volatile("global_load_async_to_lds_b128 %0, %1, off"
                         :: "v"(l + i * 4096), "v"(g + i * 4096)
                         : "memory");
        }
    };

    issueTile(0, 0);   // prologue DMA for tile 0

    // ---- Load this wave's 16-row A tile, split to bf16 hi/lo in registers.
    // A k order used here (contiguous, same as B): lane<16 : K = {0..15};
    //                          lane>=16: K = {16..31}; row M = lane&15.
    const int rowBase = blockIdx.x * 128 + wave * 16;
    const float* xr = x + (size_t)(rowBase + col) * D_DIM;

    BF16x16 Ahi[DC], Alo[DC];
#pragma unroll
    for (int dc = 0; dc < DC; ++dc) {
        int d0 = dc * 32 + half * 16;
#pragma unroll
        for (int i = 0; i < 8; ++i) {
            unsigned short h0, l0, h1, l1;
            split_bf16(xr[d0 + i],      h0, l0);
            split_bf16(xr[d0 + 8 + i],  h1, l1);
            Ahi[dc].u[i]     = h0;  Alo[dc].u[i]     = l0;
            Ahi[dc].u[8 + i] = h1;  Alo[dc].u[8 + i] = l1;
        }
    }

    float best[8];
    int   bidx[8];
#pragma unroll
    for (int j = 0; j < 8; ++j) { best[j] = 3.4e38f; bidx[j] = 0; }

    for (int kt = 0; kt < KT; ++kt) {
        const int buf = kt & 1;
        // tile kt finished (this wave's DMA; loads complete in order)
        asm volatile("s_wait_asynccnt 0" ::: "memory");
        // all waves: tile kt visible, and everyone done reading buf^1
        __syncthreads();
        // overlap next tile's DMA with this tile's 24 WMMAs
        if (kt + 1 < KT) issueTile(kt + 1, buf ^ 1);

        v8f acc = {};
        const unsigned short* sb = s_b[buf];
#pragma unroll
        for (int dc = 0; dc < DC; ++dc) {
            BF16x16 bhi, blo;
            int c0 = (dc * 2 + 0) * 2;
            int c1 = (dc * 2 + 1) * 2;
            bhi.h[0] = *(const v8u*)(sb + ((size_t)(c0 + 0) * 32 + lane) * 8);
            bhi.h[1] = *(const v8u*)(sb + ((size_t)(c0 + 1) * 32 + lane) * 8);
            blo.h[0] = *(const v8u*)(sb + ((size_t)(c1 + 0) * 32 + lane) * 8);
            blo.h[1] = *(const v8u*)(sb + ((size_t)(c1 + 1) * 32 + lane) * 8);
            acc = __builtin_amdgcn_wmma_f32_16x16x32_bf16(
                      false, Ahi[dc].bf, false, bhi.bf, (short)0, acc, false, false);
            acc = __builtin_amdgcn_wmma_f32_16x16x32_bf16(
                      false, Ahi[dc].bf, false, blo.bf, (short)0, acc, false, false);
            acc = __builtin_amdgcn_wmma_f32_16x16x32_bf16(
                      false, Alo[dc].bf, false, bhi.bf, (short)0, acc, false, false);
        }

        // partial-distance argmin update (k strictly increasing -> '<'
        // keeps first occurrence within this lane's column stream)
        int   kcol = kt * 16 + col;
        float ek   = s_esq[kcol];
#pragma unroll
        for (int j = 0; j < 8; ++j) {
            float score = ek - 2.0f * acc[j];
            if (score < best[j]) { best[j] = score; bidx[j] = kcol; }
        }
    }

    // ---- reduce across the 16 columns (lanes within this half), tie -> min k
#pragma unroll
    for (int j = 0; j < 8; ++j) {
        float bv = best[j];
        int   bi = bidx[j];
        for (int m = 1; m <= 8; m <<= 1) {
            float ov = __shfl_xor(bv, m, 32);
            int   oi = __shfl_xor(bi, m, 32);
            if (ov < bv || (ov == bv && oi < bi)) { bv = ov; bi = oi; }
        }
        if (col == 0) codes[rowBase + half * 8 + j] = bi;
    }
}

// ---------------------------------------------------------------------------
// Kernel 3: quantized = emb[codes]; also emit codes (int32) after quantized.
// ---------------------------------------------------------------------------
__global__ void vq_gather(const float* __restrict__ emb,
                          const int* __restrict__ codes,
                          float* __restrict__ outq,
                          int* __restrict__ outcodes) {
    int n = blockIdx.x;
    int c = codes[n];
    outq[(size_t)n * D_DIM + threadIdx.x] = emb[(size_t)c * D_DIM + threadIdx.x];
    if (threadIdx.x == 0) outcodes[n] = c;
}

// ---------------------------------------------------------------------------
extern "C" void kernel_launch(void* const* d_in, const int* in_sizes, int n_in,
                              void* d_out, int out_size, void* d_ws, size_t ws_size,
                              hipStream_t stream) {
    (void)in_sizes; (void)n_in; (void)out_size; (void)ws_size;
    const float* x     = (const float*)d_in[0];   // [32,1024,256]
    const float* esum  = (const float*)d_in[1];   // [2048,256]
    const float* usage = (const float*)d_in[2];   // [2048]

    // workspace layout (~4.33 MB total)
    char* ws = (char*)d_ws;
    float*          emb   = (float*)(ws);                              // 2 MB
    float*          esq   = (float*)(ws + 2097152);                    // 8 KB
    unsigned short* btile = (unsigned short*)(ws + 2097152 + 8192);    // 2 MB
    int*            codes = (int*)(ws + 2097152 + 8192 + 2097152);     // 128 KB

    float* outq     = (float*)d_out;                                  // [B,T,D] f32
    int*   outcodes = (int*)((float*)d_out + (size_t)N_ROWS * D_DIM); // [B,T] i32

    vq_prep  <<<K_DIM,        256, 0, stream>>>(esum, usage, emb, esq, btile);
    vq_argmin<<<N_ROWS / 128, 256, 0, stream>>>(x, btile, esq, codes);
    vq_gather<<<N_ROWS,       256, 0, stream>>>(emb, codes, outq, outcodes);
}
